// InterPartMR_6356551598736
// MI455X (gfx1250) — matrix operand order repaired and checked by reading, not compiled, not run on hardware
//
#include <hip/hip_runtime.h>

// MI455X / gfx1250: wave32, WMMA 16x16x32 f16 -> f32.
// Problem is memory-bound (42 FLOP/B << f16 machine balance at 23.3 TB/s),
// so: stats pass + recompute pass (x and weights stay in 192MB L2), f16 WMMA.

typedef __attribute__((ext_vector_type(16))) _Float16 v16h;
typedef __attribute__((ext_vector_type(8)))  _Float16 v8h;
typedef __attribute__((ext_vector_type(8)))  float    v8f;

#define B_TOT   2048
#define C_IN    256
#define P_J     15
#define TWO_C   512
#define GC      128          // channels per group (in and out)
#define BT      16           // batches per block -> 240 columns = 15 N-tiles
#define NTILES  15
#define STRIDEH 248          // LDS row stride in halves (padded vs 240)
#define NBLK    (B_TOT / BT) // 128 batch tiles
#define NSPAT   (B_TOT * P_J)

// ---------------- weight fp32 -> fp16 conversion ----------------
__global__ void prep_kernel(const float* __restrict__ w, _Float16* __restrict__ wf16) {
    int i = blockIdx.x * blockDim.x + threadIdx.x;
    if (i < TWO_C * GC) wf16[i] = (_Float16)w[i];
}

// ---------------- fused stage + grouped GEMM ----------------
// STATS=true : accumulate per-channel sum / sumsq of z = Wy + b into per-block partials
// STATS=false: apply out = relu(acc*scale + bias) and store
template <bool STATS>
__global__ __launch_bounds__(256)
void gemm_kernel(const float* __restrict__ x,
                 const _Float16* __restrict__ wf16,
                 const float* __restrict__ conv_b,
                 float* __restrict__ psum,   // [512][NBLK]
                 float* __restrict__ psq,    // [512][NBLK]
                 const float* __restrict__ sb, // [2*512] scale | bias
                 float* __restrict__ out) {
    __shared__ __align__(16) _Float16 ytile[GC * STRIDEH];

    const int g     = blockIdx.y;        // group 0..3
    const int bBase = blockIdx.x * BT;   // 16 batches per block

    // ---- stage Y_g (K=128 x N=240) into LDS as f16 ----
    // groups 0/1: y = xs[ch];  groups 2/3: y = maskedmax(xs) - xs
    for (int task = threadIdx.x; task < GC * BT; task += blockDim.x) {
        const int bl  = task >> 7;        // local batch 0..15
        const int row = task & (GC - 1);  // K-row 0..127
        const int cg  = ((g & 1) * GC) + row;
        const float* src = x + ((size_t)(bBase + bl) * C_IN + cg) * P_J;
        float v[P_J];
        #pragma unroll
        for (int p = 0; p < P_J; ++p) v[p] = src[p];
        _Float16* dst = &ytile[row * STRIDEH + bl * P_J];
        if (g < 2) {
            #pragma unroll
            for (int p = 0; p < P_J; ++p) dst[p] = (_Float16)v[p];
        } else {
            float pm[5];
            #pragma unroll
            for (int q = 0; q < 5; ++q)
                pm[q] = fmaxf(fmaxf(v[3*q], v[3*q+1]), v[3*q+2]);
            float m1 = -3.0e38f, m2 = -3.0e38f;   // top-2 of part maxes
            #pragma unroll
            for (int q = 0; q < 5; ++q) {
                if (pm[q] > m1)      { m2 = m1; m1 = pm[q]; }
                else if (pm[q] > m2) { m2 = pm[q]; }
            }
            #pragma unroll
            for (int q = 0; q < 5; ++q) {
                const float mo = (pm[q] < m1) ? m1 : m2;         // max over other parts
                const float mq = fmaxf(mo, pm[q] - 10000.0f);    // exact reference semantics
                #pragma unroll
                for (int j = 0; j < 3; ++j)
                    dst[3*q + j] = (_Float16)(mq - v[3*q + j]);
            }
        }
    }
    __syncthreads();

    // ---- WMMA: Z[128 x 240] = W_g[128x128] * Y[128x240] ----
    const int wave  = threadIdx.x >> 5;   // 0..7 -> M-tile
    const int lane  = threadIdx.x & 31;
    const int mLane = lane & 15;
    const int kOff  = (lane < 16) ? 0 : 8;   // A-fragment K phase (ISA 16-bit A layout)
    const int moff  = (lane < 16) ? 0 : 8;   // C/D: upper lanes hold M=8..15
    const int ch0   = g * GC + wave * 16;    // first output channel of this wave

    // A fragments (weights), kept in registers across all 15 N-tiles
    v16h a[4];
    {
        const _Float16* wrow = wf16 + (size_t)(ch0 + mLane) * GC;
        #pragma unroll
        for (int c = 0; c < 4; ++c) {
            v8h lo = *(const v8h*)(wrow + c * 32 + kOff);
            v8h hi = *(const v8h*)(wrow + c * 32 + 16 + kOff);
            a[c] = __builtin_shufflevector(lo, hi, 0,1,2,3,4,5,6,7,8,9,10,11,12,13,14,15);
        }
    }

    float p0[8], p1[8];   // STATS: sum / sumsq ; OUT: scale / bias
    float cb[8];
    if (STATS) {
        #pragma unroll
        for (int r = 0; r < 8; ++r) { p0[r] = 0.0f; p1[r] = 0.0f; cb[r] = conv_b[ch0 + moff + r]; }
    } else {
        #pragma unroll
        for (int r = 0; r < 8; ++r) {
            const int ch = ch0 + moff + r;
            p0[r] = sb[ch];
            p1[r] = sb[TWO_C + ch];
        }
    }

    for (int nt = 0; nt < NTILES; ++nt) {
        v8f acc = {};
        #pragma unroll
        for (int c = 0; c < 4; ++c) {
            // B fragment: lane holds N = nt*16 + mLane, K = c*32 + kOff + {0..7, 16..23}
            const _Float16* yb = &ytile[(c * 32 + kOff) * STRIDEH + nt * 16 + mLane];
            v16h bb;
            #pragma unroll
            for (int s = 0; s < 8; ++s) {
                bb[s]     = yb[s * STRIDEH];
                bb[8 + s] = yb[(16 + s) * STRIDEH];
            }
            acc = __builtin_amdgcn_wmma_f32_16x16x32_f16(false, a[c], false, bb,
                                                         (short)0, acc, false, false);
        }
        if (STATS) {
            #pragma unroll
            for (int r = 0; r < 8; ++r) {
                const float z = acc[r] + cb[r];
                p0[r] += z;
                p1[r] += z * z;
            }
        } else {
            const int n  = nt * 16 + mLane;   // column within block
            const int bl = n / P_J;
            const int p  = n - bl * P_J;
            float* dst = out + ((size_t)(bBase + bl) * TWO_C + ch0 + moff) * P_J + p;
            #pragma unroll
            for (int r = 0; r < 8; ++r)
                dst[r * P_J] = fmaxf(0.0f, acc[r] * p0[r] + p1[r]);
        }
    }

    if (STATS) {
        // reduce across the 16 N-lanes of each half-wave (butterfly), then
        // write deterministic per-block partials (no float atomics)
        #pragma unroll
        for (int mask = 1; mask < 16; mask <<= 1) {
            #pragma unroll
            for (int r = 0; r < 8; ++r) {
                p0[r] += __shfl_xor(p0[r], mask, 32);
                p1[r] += __shfl_xor(p1[r], mask, 32);
            }
        }
        if (mLane == 0) {
            #pragma unroll
            for (int r = 0; r < 8; ++r) {
                const int ch = ch0 + moff + r;
                psum[(size_t)ch * NBLK + blockIdx.x] = p0[r];
                psq [(size_t)ch * NBLK + blockIdx.x] = p1[r];
            }
        }
    }
}

// ---------------- fold partials -> per-channel scale/bias ----------------
__global__ void finalize_kernel(const float* __restrict__ psum,
                                const float* __restrict__ psq,
                                const float* __restrict__ conv_b,
                                const float* __restrict__ gamma,
                                const float* __restrict__ beta,
                                float* __restrict__ sb) {
    const int c = threadIdx.x;   // 512 channels
    float s = 0.0f, q = 0.0f;
    for (int bkt = 0; bkt < NBLK; ++bkt) {
        s += psum[(size_t)c * NBLK + bkt];
        q += psq [(size_t)c * NBLK + bkt];
    }
    const float inv_n = 1.0f / (float)NSPAT;
    const float mean  = s * inv_n;
    const float var   = q * inv_n - mean * mean;
    const float scale = gamma[c] * rsqrtf(var + 1e-5f);
    sb[c]         = scale;
    sb[TWO_C + c] = beta[c] + (conv_b[c] - mean) * scale;  // folds conv bias + BN shift
}

extern "C" void kernel_launch(void* const* d_in, const int* in_sizes, int n_in,
                              void* d_out, int out_size, void* d_ws, size_t ws_size,
                              hipStream_t stream) {
    const float* x      = (const float*)d_in[0];
    const float* conv_w = (const float*)d_in[1];
    const float* conv_b = (const float*)d_in[2];
    const float* gamma  = (const float*)d_in[3];
    const float* beta   = (const float*)d_in[4];
    float* out = (float*)d_out;

    // workspace layout
    char* ws = (char*)d_ws;
    _Float16* wf16 = (_Float16*)ws;                             // 65536 halves = 128 KiB
    float* psum = (float*)(ws + 131072);                        // 512*128 f = 256 KiB
    float* psq  = (float*)(ws + 131072 + 262144);               // 512*128 f = 256 KiB
    float* sb   = (float*)(ws + 131072 + 2 * 262144);           // 1024 f

    prep_kernel<<<(TWO_C * GC + 255) / 256, 256, 0, stream>>>(conv_w, wf16);

    dim3 grid(NBLK, 4);  // 128 batch tiles x 4 conv groups
    gemm_kernel<true><<<grid, 256, 0, stream>>>(x, wf16, conv_b, psum, psq, nullptr, nullptr);
    finalize_kernel<<<1, TWO_C, 0, stream>>>(psum, psq, conv_b, gamma, beta, sb);
    gemm_kernel<false><<<grid, 256, 0, stream>>>(x, wf16, conv_b, nullptr, nullptr, sb, out);
}
